// VQAttentionOld_35974646071752
// MI455X (gfx1250) — matrix operand order repaired and checked by reading, not compiled, not run on hardware
//
#include <hip/hip_runtime.h>
#include <math.h>

// ---- problem constants (from reference) ----
#define B_ 8
#define L_ 512
#define D_ 1024
#define H_ 8
#define K_ 128
#define V_ 128
#define S_ 512
#define W_ 1024            // 2*L
#define INFTY_ 1e30f

// output layout (flat f32): res | l_commit | l_codebook | new_lower | new_upper
#define O_RES    0
#define O_COMMIT ((size_t)B_*L_*D_)
#define O_CODEBK (O_COMMIT + 1)
#define O_LOWER  (O_COMMIT + 2)
#define O_UPPER  (O_LOWER + (size_t)B_*H_*S_)

typedef __attribute__((ext_vector_type(16))) __bf16 v16bf;
typedef __attribute__((ext_vector_type(8)))  float  v8f;
typedef __attribute__((ext_vector_type(4)))  unsigned int v4u;
typedef __attribute__((ext_vector_type(8)))  int v8i;
typedef __attribute__((ext_vector_type(4)))  int v4i;

union bfu { __bf16 b; unsigned short u; };
__device__ __forceinline__ __bf16 f2bf(float f) {
  unsigned int x = __float_as_uint(f);
  unsigned int r = (x + 0x7FFFu + ((x >> 16) & 1u)) >> 16;
  bfu t; t.u = (unsigned short)r; return t.b;
}
__device__ __forceinline__ float bf2f(__bf16 b) {
  bfu t; t.b = b; return __uint_as_float(((unsigned int)t.u) << 16);
}

union fragU { uint4 q[2]; v16bf v; };

// =====================================================================
// Tensor Data Mover compile-path probe (CDNA5 async tensor DMA).
// Not launched by kernel_launch — emitted into the object so the
// tensor_load_to_lds / s_wait_tensorcnt path is exercised for gfx1250.
// This toolchain uses the 6-arg builtin form:
//   (uint32x4 g0, int32x8 g1, int32x4 g2, int32x4 g3, int32x8 pad, i32 cpol)
// =====================================================================
#if __has_builtin(__builtin_amdgcn_tensor_load_to_lds)
__global__ __launch_bounds__(32)
void tdm_probe_kernel(const float* __restrict__ src, float* __restrict__ dst)
{
  __shared__ float stage[128 * 32];
  unsigned long long ga = (unsigned long long)(size_t)src;
  // D# group0: count=1, lds_addr=0, global_addr=src, type=2 ("image")
  v4u g0;
  g0.x = 1u;
  g0.y = 0u;
  g0.z = (unsigned int)(ga & 0xffffffffu);
  g0.w = (unsigned int)((ga >> 32) & 0x01ffffffu) | (2u << 30);
  // D# group1: mask=0, data_size=4B, 2D 128x32 tile of a 1024-stride tensor
  v8i g1;
  g1[0] = (2 << 16);                 // data_size code 2 = 4 bytes
  g1[1] = (1024 << 16);              // tensor_dim0 (low bits)
  g1[2] = 0;
  g1[3] = 128 | (32 << 16);          // tile_dim0 / tile_dim1
  g1[4] = 32;                        // tile_dim2 region
  g1[5] = 1024;                      // tensor_dim0_stride (low bits)
  g1[6] = 0;
  g1[7] = 0;
  v4i gz4 = {0, 0, 0, 0};
  v8i gz8 = {0, 0, 0, 0, 0, 0, 0, 0};
  __builtin_amdgcn_tensor_load_to_lds(g0, g1, gz4, gz4, gz8, 0);
  __builtin_amdgcn_s_wait_tensorcnt((short)0);
  __syncthreads();
  for (int i = threadIdx.x; i < 128 * 32; i += 32) dst[i] = stage[i];
}
#endif

// =====================================================================
// Generic bf16 WMMA GEMM: C = alpha * A @ op(B) + beta * C
// A: [M,K] row-major (lda), B: NN -> [K,N] (ldb) ; NT -> [N,K] (ldb)
// Optional f32 output C and/or bf16 output Cbf. batch via blockIdx.z.
// Tile 128x128, K-step 32. 8 waves -> each wave 32x64 = 2x4 WMMA frags.
// LDS tiles are both [rows][32] with 96B row stride so every fragment
// load is a pair of 16B ds_load_b128.
// =====================================================================
#define LDT_S 48    // 96B row stride (multiple of 16B)

template<bool TB>
__global__ __launch_bounds__(256)
void gemm_bf16_kernel(const __bf16* __restrict__ A, long lda, long sAb,
                      const __bf16* __restrict__ Bm, long ldb, long sBb,
                      float* __restrict__ C, long ldc, long sCb,
                      __bf16* __restrict__ Cbf, long ldcbf, long sCbfb,
                      int M, int N, int Kd, float alpha, float beta)
{
  __shared__ __bf16 As[128][LDT_S];   // A tile: [m][k]
  __shared__ __bf16 Bs[128][LDT_S];   // B tile: [n][k]  (fragment friendly)

  const int batch = blockIdx.z;
  A  += (size_t)batch * sAb;
  Bm += (size_t)batch * sBb;
  float* Cb = C ? C + (size_t)batch * sCb : nullptr;
  __bf16* Cf = Cbf ? Cbf + (size_t)batch * sCbfb : nullptr;

  const int tn0 = blockIdx.x * 128;
  const int tm0 = blockIdx.y * 128;
  const int tid  = threadIdx.x;
  const int lane = tid & 31;
  const int wid  = tid >> 5;
  const int wm = wid >> 1;   // 0..3
  const int wn = wid & 1;    // 0..1

  const v8f zv = {0.f,0.f,0.f,0.f,0.f,0.f,0.f,0.f};
  v8f acc[2][4];
#pragma unroll
  for (int i=0;i<2;i++)
#pragma unroll
    for (int j=0;j<4;j++) acc[i][j] = zv;

  for (int k0 = 0; k0 < Kd; k0 += 32) {
    if (k0 + 32 < Kd) {   // hint next tiles (global_prefetch_b8)
      __builtin_prefetch(A + (size_t)tm0*lda + k0 + 32, 0, 0);
      __builtin_prefetch(Bm + (TB ? (size_t)tn0*ldb + k0 + 32
                                  : (size_t)(k0+32)*ldb + tn0), 0, 0);
    }
    // ---- A tile: 128 x 32 bf16, 512 16B vectors, 2 per thread ----
#pragma unroll
    for (int t = 0; t < 2; ++t) {
      int vec = tid + t*256;
      int row = vec >> 2;
      int c8  = (vec & 3) * 8;
      uint4 d = *(const uint4*)(A + (size_t)(tm0 + row)*lda + k0 + c8);
      *(uint4*)(&As[row][c8]) = d;
    }
    // ---- B tile into Bs[n][k] ----
    if (TB) {
      // B is [N,K] row-major: straight contiguous copy
#pragma unroll
      for (int t = 0; t < 2; ++t) {
        int vec = tid + t*256;
        int nn  = vec >> 2;
        int k8  = (vec & 3) * 8;
        uint4 d = *(const uint4*)(Bm + (size_t)(tn0 + nn)*ldb + k0 + k8);
        *(uint4*)(&Bs[nn][k8]) = d;
      }
    } else {
      // B is [K,N] row-major: transpose while staging
#pragma unroll
      for (int t = 0; t < 2; ++t) {
        int vec = tid + t*256;
        int kk  = vec >> 4;
        int c8  = (vec & 15) * 8;
        uint4 d = *(const uint4*)(Bm + (size_t)(k0 + kk)*ldb + tn0 + c8);
        const __bf16* e = (const __bf16*)&d;
#pragma unroll
        for (int i = 0; i < 8; i++) Bs[c8 + i][kk] = e[i];
      }
    }
    __syncthreads();

    // ---- A fragments: lane m = lane&15; lanes0-15 K{0..7,16..23},
    //      lanes16-31 K{8..15,24..31} -> two ds_load_b128 each ----
    v16bf af[2];
    {
      int kb = (lane < 16) ? 0 : 8;
#pragma unroll
      for (int i=0;i<2;i++) {
        int row = wm*32 + i*16 + (lane & 15);
        fragU u;
        u.q[0] = *(const uint4*)(&As[row][kb]);
        u.q[1] = *(const uint4*)(&As[row][kb + 16]);
        af[i] = u.v;
      }
    }
    // ---- B fragments: lane n = lane&15; lanes0-15 K{0..7,16..23},
    //      lanes16-31 K{8..15,24..31} -> two ds_load_b128 each ----
    v16bf bfm[4];
    {
      int kb = (lane < 16) ? 0 : 8;
#pragma unroll
      for (int j=0;j<4;j++) {
        int col = wn*64 + j*16 + (lane & 15);
        fragU u;
        u.q[0] = *(const uint4*)(&Bs[col][kb]);
        u.q[1] = *(const uint4*)(&Bs[col][kb + 16]);
        bfm[j] = u.v;
      }
    }
#pragma unroll
    for (int i=0;i<2;i++)
#pragma unroll
      for (int j=0;j<4;j++)
        acc[i][j] = __builtin_amdgcn_wmma_f32_16x16x32_bf16(
            false, af[i], false, bfm[j], (short)0, acc[i][j], false, false);
    __syncthreads();
  }

  // ---- store (C/D layout: lanes0-15 m=r, lanes16-31 m=8+r, n=lane&15) ----
  const int cn = lane & 15;
  const int rb = (lane < 16) ? 0 : 8;
#pragma unroll
  for (int i=0;i<2;i++) {
#pragma unroll
    for (int j=0;j<4;j++) {
      int col  = tn0 + wn*64 + j*16 + cn;
      int mrow = tm0 + wm*32 + i*16 + rb;
#pragma unroll
      for (int r=0;r<8;r++) {
        int row = mrow + r;
        float v = alpha * acc[i][j][r];
        if (beta != 0.0f) v += beta * Cb[(size_t)row*ldc + col];
        if (Cb) Cb[(size_t)row*ldc + col] = v;
        if (Cf) Cf[(size_t)row*ldcbf + col] = f2bf(v);
      }
    }
  }
}

// =====================================================================
// Elementwise / setup kernels
// =====================================================================
__global__ __launch_bounds__(256)
void rmsnorm_kernel(const float* __restrict__ x, const float* __restrict__ g,
                    __bf16* __restrict__ xt)
{
  int row = blockIdx.x;
  const float* xr = x + (size_t)row * D_;
  float ss = 0.f;
  for (int i = threadIdx.x; i < D_; i += 256) { float v = xr[i]; ss += v*v; }
#pragma unroll
  for (int m = 16; m >= 1; m >>= 1) ss += __shfl_xor(ss, m, 32);
  __shared__ float red[8];
  if ((threadIdx.x & 31) == 0) red[threadIdx.x >> 5] = ss;
  __syncthreads();
  float tot = 0.f;
#pragma unroll
  for (int i=0;i<8;i++) tot += red[i];
  float sc = rsqrtf(tot / (float)D_ + 1e-6f);
  for (int i = threadIdx.x; i < D_; i += 256)
    xt[(size_t)row*D_ + i] = f2bf(xr[i] * sc * g[i]);
}

__global__ void f2bf_kernel(const float* __restrict__ in, __bf16* __restrict__ out, long n)
{
  long i = (long)blockIdx.x*256 + threadIdx.x;
  if (i < n) out[i] = f2bf(in[i]);
}

__global__ void sinusoid_kernel(__bf16* __restrict__ P)
{
  long idx = (long)blockIdx.x*256 + threadIdx.x;   // W_*D_ elements
  int j = (int)(idx & (D_-1));
  int w = (int)(idx >> 10);
  float pos = (float)(W_ - 1 - w);
  int i = (j < D_/2) ? j : j - D_/2;
  float freq = __expf(-(2.0f * (float)i / (float)D_) * 9.210340371976184f); // ln(1e4)
  float ang = pos * freq;
  float v = (j < D_/2) ? __sinf(ang) : __cosf(ang);
  P[idx] = f2bf(v);
}

__global__ void qbias_kernel(const float* __restrict__ q32,
                             const float* __restrict__ xlu, const float* __restrict__ xlv,
                             __bf16* __restrict__ qu, __bf16* __restrict__ qv)
{
  long idx = (long)blockIdx.x*256 + threadIdx.x;   // B*L*H*K
  int kk = (int)(idx & 127);
  int h  = (int)((idx >> 7) & 7);
  float q = q32[idx];
  qu[idx] = f2bf(q + xlu[h*K_ + kk]);
  qv[idx] = f2bf(q + xlv[h*K_ + kk]);
}

__global__ void zero_losses_kernel(float* __restrict__ out)
{
  if (threadIdx.x == 0) { out[O_COMMIT] = 0.f; out[O_CODEBK] = 0.f; }
}

// =====================================================================
// VQ: per (b,h,l) argmin_s ||k - c_s||^2 ; writes k_hat (bf16) + losses
// =====================================================================
__global__ __launch_bounds__(128)
void vq_kernel(const float* __restrict__ k32 /*[B,L,H,K]*/,
               const float* __restrict__ codebook /*[H,S,K]*/,
               __bf16* __restrict__ khat_bf /*[B,L,H,K]*/,
               float* __restrict__ out /*d_out base*/)
{
  __shared__ __bf16 kt[128][K_];   // 32KB
  __shared__ __bf16 cb[64][K_];    // 16KB
  __shared__ float  cn2[64];
  __shared__ float  red[4];
  int l = blockIdx.x * 128 + threadIdx.x;
  int h = blockIdx.y, b = blockIdx.z;
  int tid = threadIdx.x;

  const float* krow = k32 + ((size_t)((b*L_ + l)*H_ + h)) * K_;
  float knorm = 0.f;
  for (int kk = 0; kk < K_; ++kk) { float v = krow[kk]; kt[tid][kk] = f2bf(v); knorm += v*v; }

  float best = 3.0e38f; int bests = 0;
  const float* cbh = codebook + (size_t)h * S_ * K_;
  for (int c0 = 0; c0 < S_; c0 += 64) {
    __syncthreads();
    for (int t = tid; t < 64*K_; t += 128) {
      int s = t >> 7, kk = t & 127;
      cb[s][kk] = f2bf(cbh[(size_t)(c0 + s)*K_ + kk]);
    }
    __syncthreads();
    if (tid < 64) {
      float cs = 0.f;
      for (int kk = 0; kk < K_; ++kk) { float v = bf2f(cb[tid][kk]); cs += v*v; }
      cn2[tid] = cs;
    }
    __syncthreads();
    for (int s = 0; s < 64; ++s) {
      float dot = 0.f;
      for (int kk = 0; kk < K_; ++kk) dot += bf2f(kt[tid][kk]) * bf2f(cb[s][kk]);
      float d2 = knorm - 2.f*dot + cn2[s];
      if (d2 < best) { best = d2; bests = c0 + s; }
    }
  }
  const float* crow = cbh + (size_t)bests * K_;
  __bf16* kout = khat_bf + ((size_t)((b*L_ + l)*H_ + h)) * K_;
  for (int kk = 0; kk < K_; ++kk) kout[kk] = f2bf(crow[kk]);

  float s = best;
#pragma unroll
  for (int m = 16; m >= 1; m >>= 1) s += __shfl_xor(s, m, 32);
  if ((tid & 31) == 0) red[tid >> 5] = s;
  __syncthreads();
  if (tid == 0) {
    float tot = red[0]+red[1]+red[2]+red[3];
    float scale = 1.0f / (float)(B_*H_*L_);
    atomicAdd(&out[O_COMMIT], tot * scale);
    atomicAdd(&out[O_CODEBK], tot * scale);
  }
}

// =====================================================================
// Per-b helpers: build recent window, combine bd+mask, softmax
// =====================================================================
__global__ void rkrv_kernel(const float* __restrict__ xlckh, const float* __restrict__ xlcv,
                            const __bf16* __restrict__ khat_bf, const __bf16* __restrict__ v_bf,
                            __bf16* __restrict__ rk, __bf16* __restrict__ rv, int b)
{
  long idx = (long)blockIdx.x*256 + threadIdx.x;     // H*W*128
  int kk = (int)(idx & 127);
  int w  = (int)((idx >> 7) & (W_-1));
  int h  = (int)(idx >> 17);
  __bf16 kv, vv;
  if (w < L_) {
    size_t src = ((size_t)((b*H_ + h)*L_ + w))*K_ + kk;
    kv = f2bf(xlckh[src]);
    vv = f2bf(xlcv[src]);
  } else {
    size_t src = ((size_t)((b*L_ + (w - L_))*H_ + h))*K_ + kk;
    kv = khat_bf[src];
    vv = v_bf[src];
  }
  rk[idx] = kv; rv[idx] = vv;
}

__global__ void combine_kernel(float* __restrict__ sr, const float* __restrict__ bdraw,
                               const int* __restrict__ pos_offset)
{
  long idx = (long)blockIdx.x*256 + threadIdx.x;     // H*L*W
  int w = (int)(idx & (W_-1));
  int l = (int)((idx >> 10) & (L_-1));
  int h = (int)(idx >> 19);
  int il = L_ - pos_offset[0]; if (il < 0) il = 0;
  float s = sr[idx];
  if (w < il || w > l + L_) s = -INFTY_;
  else if (w > l) s += bdraw[((size_t)(h*L_ + l))*L_ + (w - l - 1)];
  sr[idx] = s;
}

__global__ __launch_bounds__(256)
void softmax_kernel(const float* __restrict__ sr, const float* __restrict__ sc,
                    const float* __restrict__ agg_lower_b,
                    __bf16* __restrict__ pr, __bf16* __restrict__ pc)
{
  int l = blockIdx.x, h = blockIdx.y;
  __shared__ float vals[W_ + S_];
  __shared__ float red[8];
  const float* r = sr + ((size_t)h*L_ + l)*W_;
  const float* c = sc + ((size_t)h*L_ + l)*S_;
  const float* low = agg_lower_b + (size_t)h*S_;
  float mx = -3.0e38f;
  for (int i = threadIdx.x; i < W_+S_; i += 256) {
    float v;
    if (i < W_) v = r[i];
    else {
      int s = i - W_;
      float lw = low[s];
      float bias = (lw == 0.f) ? -INFTY_ : __logf(fmaxf(lw, 1.f));
      v = c[s] + bias;
    }
    vals[i] = v;
    mx = fmaxf(mx, v);
  }
#pragma unroll
  for (int m = 16; m >= 1; m >>= 1) mx = fmaxf(mx, __shfl_xor(mx, m, 32));
  if ((threadIdx.x & 31) == 0) red[threadIdx.x >> 5] = mx;
  __syncthreads();
  mx = red[0];
#pragma unroll
  for (int i = 1; i < 8; i++) mx = fmaxf(mx, red[i]);

  float sum = 0.f;
  for (int i = threadIdx.x; i < W_+S_; i += 256) {
    float e = __expf(vals[i] - mx);
    vals[i] = e;
    sum += e;
  }
  __syncthreads();
#pragma unroll
  for (int m = 16; m >= 1; m >>= 1) sum += __shfl_xor(sum, m, 32);
  if ((threadIdx.x & 31) == 0) red[threadIdx.x >> 5] = sum;
  __syncthreads();
  float tot = 0.f;
#pragma unroll
  for (int i = 0; i < 8; i++) tot += red[i];
  float inv = 1.0f / tot;

  __bf16* po = pr + ((size_t)h*L_ + l)*W_;
  __bf16* co = pc + ((size_t)h*L_ + l)*S_;
  for (int i = threadIdx.x; i < W_+S_; i += 256) {
    float p = vals[i] * inv;
    if (i < W_) po[i] = f2bf(p); else co[i - W_] = f2bf(p);
  }
}

// =====================================================================
// State update (uses only xlc_z / xlc_v — evicted half of the window)
// =====================================================================
__global__ __launch_bounds__(256)
void lower_kernel(const float* __restrict__ agg_lower, const int* __restrict__ xlcz,
                  float* __restrict__ out)
{
  int bh = blockIdx.x;
  __shared__ int cnt[S_];
  for (int i = threadIdx.x; i < S_; i += 256) cnt[i] = 0;
  __syncthreads();
  const int* z = xlcz + (size_t)bh * L_;
  for (int i = threadIdx.x; i < L_; i += 256) atomicAdd(&cnt[z[i]], 1);
  __syncthreads();
  for (int i = threadIdx.x; i < S_; i += 256)
    out[O_LOWER + (size_t)bh*S_ + i] = agg_lower[(size_t)bh*S_ + i] + (float)cnt[i];
}

__global__ __launch_bounds__(128)
void upper_kernel(const float* __restrict__ agg_lower, const float* __restrict__ agg_upper,
                  const int* __restrict__ xlcz, const float* __restrict__ xlcv,
                  float* __restrict__ out)
{
  int bh = blockIdx.x;
  int v = threadIdx.x;                 // 0..127, each thread owns column v
  __shared__ float acc[64][V_];        // 32KB
  const int* z = xlcz + (size_t)bh*L_;
  const float* xv = xlcv + (size_t)bh*L_*V_;
  for (int c0 = 0; c0 < S_; c0 += 64) {
    for (int s = 0; s < 64; ++s) acc[s][v] = 0.f;
    __syncthreads();
    for (int l = 0; l < L_; ++l) {
      int s = z[l];
      if (s >= c0 && s < c0 + 64) acc[s - c0][v] += xv[(size_t)l*V_ + v];
    }
    __syncthreads();
    for (int s = 0; s < 64; ++s) {
      size_t o = (size_t)bh*S_ + c0 + s;
      float nl = out[O_LOWER + o];
      float dl = fmaxf(nl, 1.f);
      float al = agg_lower[o];
      out[O_UPPER + o*V_ + v] = (al/dl)*agg_upper[o*V_ + v] + acc[s][v]/dl;
    }
    __syncthreads();
  }
}

// =====================================================================
// host side
// =====================================================================
static void gemm(bool tb,
                 const __bf16* A, long lda, long sAb,
                 const __bf16* Bm, long ldb, long sBb,
                 float* C, long ldc, long sCb,
                 __bf16* Cbf, long ldcbf, long sCbfb,
                 int M, int N, int Kd, float alpha, float beta,
                 int batch, hipStream_t st)
{
  dim3 g(N/128, M/128, batch), blk(256);
  if (tb)
    gemm_bf16_kernel<true ><<<g, blk, 0, st>>>(A,lda,sAb,Bm,ldb,sBb,C,ldc,sCb,Cbf,ldcbf,sCbfb,M,N,Kd,alpha,beta);
  else
    gemm_bf16_kernel<false><<<g, blk, 0, st>>>(A,lda,sAb,Bm,ldb,sBb,C,ldc,sCb,Cbf,ldcbf,sCbfb,M,N,Kd,alpha,beta);
}

extern "C" void kernel_launch(void* const* d_in, const int* in_sizes, int n_in,
                              void* d_out, int out_size, void* d_ws, size_t ws_size,
                              hipStream_t stream)
{
  const float* x        = (const float*)d_in[0];
  const float* w_ln     = (const float*)d_in[1];
  const float* w_q      = (const float*)d_in[2];
  const float* w_k      = (const float*)d_in[3];
  const float* w_v      = (const float*)d_in[4];
  const float* w_r      = (const float*)d_in[5];
  const float* xl_u     = (const float*)d_in[6];
  const float* xl_v     = (const float*)d_in[7];
  const float* codebook = (const float*)d_in[8];
  const float* w_res    = (const float*)d_in[9];
  const float* xlc_k_hat= (const float*)d_in[10];
  const float* xlc_v    = (const float*)d_in[11];
  const float* agg_upper= (const float*)d_in[12];
  const float* agg_lower= (const float*)d_in[13];
  const int*   xlc_z    = (const int*)d_in[14];
  const int*   pos_off  = (const int*)d_in[15];
  float* out = (float*)d_out;

  // ---- workspace carve-up ----
  char* wp = (char*)d_ws;
  auto alloc = [&](size_t bytes) -> void* {
    void* p = (void*)wp; wp += (bytes + 255) & ~(size_t)255; return p;
  };
  const size_t NT = (size_t)B_*L_;           // 4096 token rows
  __bf16* xt_bf   = (__bf16*)alloc(NT*D_*2);
  __bf16* wqb     = (__bf16*)alloc((size_t)D_*H_*K_*2);
  __bf16* wkb     = (__bf16*)alloc((size_t)D_*H_*K_*2);
  __bf16* wvb     = (__bf16*)alloc((size_t)D_*H_*V_*2);
  __bf16* wrb     = (__bf16*)alloc((size_t)D_*H_*K_*2);
  __bf16* wresb   = (__bf16*)alloc((size_t)H_*V_*D_*2);
  __bf16* Pb      = (__bf16*)alloc((size_t)W_*D_*2);
  float*  q32     = (float*) alloc(NT*H_*K_*4);
  __bf16* qu_bf   = (__bf16*)alloc(NT*H_*K_*2);
  __bf16* qv_bf   = (__bf16*)alloc(NT*H_*K_*2);
  float*  k32     = (float*) alloc(NT*H_*K_*4);
  __bf16* v_bf    = (__bf16*)alloc(NT*H_*V_*2);
  __bf16* khat_bf = (__bf16*)alloc(NT*H_*K_*2);
  __bf16* cb_bf   = (__bf16*)alloc((size_t)H_*S_*K_*2);
  __bf16* xlr_bf  = (__bf16*)alloc((size_t)W_*H_*K_*2);
  __bf16* aggup_bf= (__bf16*)alloc((size_t)B_*H_*S_*V_*2);
  __bf16* wv_bf   = (__bf16*)alloc(NT*H_*V_*2);
  // per-b reused scratch
  __bf16* rk_bf   = (__bf16*)alloc((size_t)H_*W_*K_*2);
  __bf16* rv_bf   = (__bf16*)alloc((size_t)H_*W_*V_*2);
  float*  sr32    = (float*) alloc((size_t)H_*L_*W_*4);
  float*  bdraw   = (float*) alloc((size_t)H_*L_*L_*4);
  float*  sc32    = (float*) alloc((size_t)H_*L_*S_*4);
  __bf16* pr_bf   = (__bf16*)alloc((size_t)H_*L_*W_*2);
  __bf16* pc_bf   = (__bf16*)alloc((size_t)H_*L_*S_*2);
  float*  wv32    = (float*) alloc((size_t)L_*H_*V_*4);

  const float inv_tau = 0.08838834764831845f;   // 1/sqrt(128)

  // 1) RMS norm -> bf16 tokens
  rmsnorm_kernel<<<dim3((unsigned)NT), 256, 0, stream>>>(x, w_ln, xt_bf);

  // 2) weight / constant conversions
  {
    long n = (long)D_*H_*K_;
    f2bf_kernel<<<dim3((unsigned)(n/256)), 256, 0, stream>>>(w_q, wqb, n);
    f2bf_kernel<<<dim3((unsigned)(n/256)), 256, 0, stream>>>(w_k, wkb, n);
    f2bf_kernel<<<dim3((unsigned)(n/256)), 256, 0, stream>>>(w_v, wvb, n);
    f2bf_kernel<<<dim3((unsigned)(n/256)), 256, 0, stream>>>(w_r, wrb, n);
    f2bf_kernel<<<dim3((unsigned)(n/256)), 256, 0, stream>>>(w_res, wresb, n);
    long nc = (long)H_*S_*K_;
    f2bf_kernel<<<dim3((unsigned)(nc/256)), 256, 0, stream>>>(codebook, cb_bf, nc);
    long na = (long)B_*H_*S_*V_;
    f2bf_kernel<<<dim3((unsigned)(na/256)), 256, 0, stream>>>(agg_upper, aggup_bf, na);
    sinusoid_kernel<<<dim3((unsigned)((size_t)W_*D_/256)), 256, 0, stream>>>(Pb);
  }

  // 3) projections (WMMA GEMMs)
  gemm(false, xt_bf, D_, 0, wqb, H_*K_, 0, q32, H_*K_, 0, nullptr,0,0,
       (int)NT, H_*K_, D_, inv_tau, 0.f, 1, stream);
  gemm(false, xt_bf, D_, 0, wkb, H_*K_, 0, k32, H_*K_, 0, nullptr,0,0,
       (int)NT, H_*K_, D_, 1.f, 0.f, 1, stream);
  gemm(false, xt_bf, D_, 0, wvb, H_*V_, 0, nullptr,0,0, v_bf, H_*V_, 0,
       (int)NT, H_*V_, D_, 1.f, 0.f, 1, stream);
  gemm(false, Pb, D_, 0, wrb, H_*K_, 0, nullptr,0,0, xlr_bf, H_*K_, 0,
       W_, H_*K_, D_, 1.f, 0.f, 1, stream);

  // 4) q + xl_u / q + xl_v in bf16
  qbias_kernel<<<dim3((unsigned)(NT*H_*K_/256)), 256, 0, stream>>>(q32, xl_u, xl_v, qu_bf, qv_bf);

  // 5) VQ + losses
  zero_losses_kernel<<<1, 32, 0, stream>>>(out);
  vq_kernel<<<dim3(L_/128, H_, B_), 128, 0, stream>>>(k32, codebook, khat_bf, out);

  // 6) attention, per batch element
  for (int b = 0; b < B_; ++b) {
    const long qBase = (long)b*L_*H_*K_;
    rkrv_kernel<<<dim3((unsigned)((size_t)H_*W_*K_/256)), 256, 0, stream>>>(
        xlc_k_hat, xlc_v, khat_bf, v_bf, rk_bf, rv_bf, b);
    // ac: (q/t+u) @ recent_khat^T   [per h: 512x1024, K=128]
    gemm(true, qu_bf + qBase, H_*K_, K_, rk_bf, K_, (long)W_*K_,
         sr32, W_, (long)L_*W_, nullptr,0,0, L_, W_, K_, 1.f, 0.f, H_, stream);
    // bd raw: (q/t+v) @ xl_r[512..1023]^T  [per h: 512x512]
    gemm(true, qv_bf + qBase, H_*K_, K_, xlr_bf + (long)L_*H_*K_, H_*K_, K_,
         bdraw, L_, (long)L_*L_, nullptr,0,0, L_, L_, K_, 1.f, 0.f, H_, stream);
    combine_kernel<<<dim3((unsigned)((size_t)H_*L_*W_/256)), 256, 0, stream>>>(sr32, bdraw, pos_off);
    // cache scores: (q/t+u) @ codebook^T   [per h: 512x512]
    gemm(true, qu_bf + qBase, H_*K_, K_, cb_bf, K_, (long)S_*K_,
         sc32, S_, (long)L_*S_, nullptr,0,0, L_, S_, K_, 1.f, 0.f, H_, stream);
    softmax_kernel<<<dim3(L_, H_), 256, 0, stream>>>(
        sr32, sc32, agg_lower + (size_t)b*H_*S_, pr_bf, pc_bf);
    // wv = P_recent @ recent_v  (+ P_cache @ agg_upper)
    gemm(false, pr_bf, W_, (long)L_*W_, rv_bf, V_, (long)W_*V_,
         wv32, H_*V_, V_, nullptr,0,0, L_, V_, W_, 1.f, 0.f, H_, stream);
    gemm(false, pc_bf, S_, (long)L_*S_, aggup_bf + (size_t)b*H_*S_*V_, V_, (long)S_*V_,
         wv32, H_*V_, V_, wv_bf + (size_t)b*L_*H_*V_, H_*V_, V_,
         L_, V_, S_, 1.f, 1.f, H_, stream);
  }

  // 7) output projection -> res (d_out head)
  gemm(false, wv_bf, H_*V_, 0, wresb, D_, 0, out + O_RES, D_, 0, nullptr,0,0,
       (int)NT, D_, H_*V_, 1.f, 0.f, 1, stream);

  // 8) state update
  lower_kernel<<<dim3(B_*H_), 256, 0, stream>>>(agg_lower, xlc_z, out);
  upper_kernel<<<dim3(B_*H_), 128, 0, stream>>>(agg_lower, agg_upper, xlc_z, xlc_v, out);

  (void)in_sizes; (void)n_in; (void)out_size; (void)ws_size;
}
